// CluTSPSolver_83399674953915
// MI455X (gfx1250) — compile-verified, matrix operand order checked by reading, not run on hardware
//
#include <hip/hip_runtime.h>
#include <hip/hip_bf16.h>

// ---------------------------------------------------------------------------
// Problem constants (match reference)
// ---------------------------------------------------------------------------
#define Bsz   256
#define Nn    500
#define Cc    50
#define Ee    256
#define FFd   512
#define NEGV  (-1e9f)

typedef __attribute__((ext_vector_type(16))) __bf16 v16bf;
typedef __attribute__((ext_vector_type(8)))  float  v8f;

// ---------------------------------------------------------------------------
// bf16 WMMA GEMM:  C[M,Nc] = act( A[M,K] @ W[Nc,K]^T + bias )
// One 16x16 output tile per wave32; 8 waves per 256-thread block.
// A fragment (16-bit 16x32 layout): lane l<16 holds row m0+l, K = kb+half*8+i
// (i<8) and K = kb+16+half*8+i (i>=8), half = lane>>4.  B fragment mirrors it
// over columns: lane's column n0+(lane&15) is row (n0+l) of W, contiguous in k.
// ---------------------------------------------------------------------------
template <int K>
__global__ __launch_bounds__(256)
void gemm_bf16_wmma(const float* __restrict__ A, const float* __restrict__ W,
                    const float* __restrict__ bias, float* __restrict__ C,
                    int M, int Nc, int relu)
{
    const int wave = threadIdx.x >> 5;
    const int lane = threadIdx.x & 31;
    const int hf   = lane >> 4;     // 0 or 1
    const int l    = lane & 15;

    const int tilesN = Nc >> 4;
    const int tile   = blockIdx.x * 8 + wave;
    const int total  = (M >> 4) * tilesN;
    if (tile >= total) return;

    const int m0 = (tile / tilesN) << 4;
    const int n0 = (tile % tilesN) << 4;

    const float* __restrict__ Arow = A + (size_t)(m0 + l) * K;
    const float* __restrict__ Wrow = W + (size_t)(n0 + l) * K;

    v8f acc = {};
    for (int kb = 0; kb < K; kb += 32) {
        const float* ap0 = Arow + kb + hf * 8;
        const float* ap1 = ap0 + 16;
        const float* bp0 = Wrow + kb + hf * 8;
        const float* bp1 = bp0 + 16;
        if (kb + 32 < K) {              // next-tile prefetch -> global_prefetch_b8
            __builtin_prefetch(ap0 + 32, 0, 1);
            __builtin_prefetch(bp0 + 32, 0, 1);
        }
        v16bf a, b;
#pragma unroll
        for (int i = 0; i < 8; ++i) {
            a[i]     = (__bf16)ap0[i];
            a[8 + i] = (__bf16)ap1[i];
            b[i]     = (__bf16)bp0[i];
            b[8 + i] = (__bf16)bp1[i];
        }
        acc = __builtin_amdgcn_wmma_f32_16x16x32_bf16(
                  false, a, false, b, (short)0, acc, false, false);
    }

    const int col = n0 + l;
    const float bz = bias ? bias[col] : 0.0f;
#pragma unroll
    for (int r = 0; r < 8; ++r) {
        const int row = m0 + r + hf * 8;            // C layout: VGPR r, halves
        float v = acc[r] + bz;
        if (relu) v = fmaxf(v, 0.0f);
        C[(size_t)row * Nc + col] = v;
    }
}

// ---------------------------------------------------------------------------
// S1: masked means + context concat  (one block per batch, thread = e dim)
// ctx3[b] = [unvisit_mean | current | depot]  (768)
// uclu[b] = masked mean with (mask|cluster_mask)
// ---------------------------------------------------------------------------
__global__ __launch_bounds__(256)
void build_ctx_kernel(const float* __restrict__ node_emb,
                      const float* __restrict__ cur,
                      const float* __restrict__ depot,
                      const unsigned char* __restrict__ mask,
                      const unsigned char* __restrict__ cmask,
                      float* __restrict__ ctx3,
                      float* __restrict__ uclu)
{
    const int b = blockIdx.x, e = threadIdx.x;
    const float* ne = node_emb + (size_t)b * Nn * Ee;
    const unsigned char* mk = mask  + (size_t)b * Nn;
    const unsigned char* cm = cmask + (size_t)b * Nn;
    float s0 = 0.f, s1 = 0.f;
    for (int n = 0; n < Nn; ++n) {
        const float v = ne[(size_t)n * Ee + e];
        if (!mk[n])            s0 += v;
        if (!(mk[n] | cm[n]))  s1 += v;
    }
    const float invN = 1.0f / (float)Nn;
    ctx3[(size_t)b * 768 + e]       = s0 * invN;
    ctx3[(size_t)b * 768 + 256 + e] = cur[(size_t)b * Ee + e];
    ctx3[(size_t)b * 768 + 512 + e] = depot[(size_t)b * Ee + e];
    uclu[(size_t)b * Ee + e]        = s1 * invN;
}

// ---------------------------------------------------------------------------
// S2: cluster glimpse MHA (1 query, 50 keys, 16 heads) + Wo projection +
// tanh-clipped logits + log_softmax + argmax + aug-context assembly.
// One block per batch, 256 threads.
// ---------------------------------------------------------------------------
__global__ __launch_bounds__(256)
void cluster_guidance_kernel(const float* __restrict__ cquery,   // [B,E]
                             const float* __restrict__ ckeys,    // [B,C,E]
                             const float* __restrict__ cvals,    // [B,C,E]
                             const float* __restrict__ pkeys,    // [B,C,E]
                             const unsigned char* __restrict__ vcm_in, // [B,C]
                             const unsigned char* __restrict__ is_new, // [B]
                             const float* __restrict__ Wo,       // [E,E]
                             const float* __restrict__ bo,       // [E]
                             const float* __restrict__ cluster_emb, // [B,C,E]
                             const float* __restrict__ uclu,     // [B,E]
                             const float* __restrict__ cur,
                             const float* __restrict__ depot,
                             float* __restrict__ aug,            // [B,4,E]
                             float* __restrict__ out_gemb,       // [B,E]
                             float* __restrict__ out_guid,       // [B]
                             float* __restrict__ out_prob)       // [B,C]
{
    const int b = blockIdx.x, t = threadIdx.x;
    __shared__ float sm_q[Ee];
    __shared__ float glimpse[Ee];
    __shared__ float gproj[Ee];
    __shared__ float logits[Cc];
    __shared__ unsigned char svcm[Cc];
    __shared__ float red[2];
    __shared__ int   sarg;

    sm_q[t] = cquery[(size_t)b * Ee + t];
    if (t == 0) {
        bool allv = true;
        for (int c = 1; c < Cc; ++c) allv = allv && (vcm_in[(size_t)b * Cc + c] != 0);
        const bool nb = is_new[b] != 0;
        svcm[0] = nb ? (unsigned char)(!allv) : vcm_in[(size_t)b * Cc];
    }
    if (t >= 1 && t < Cc) svcm[t] = vcm_in[(size_t)b * Cc + t];
    __syncthreads();

    {   // per-head online-softmax attention, d=16, 1 query
        const int h = t >> 4, d = t & 15;
        const float* Kp = ckeys + (size_t)b * Cc * Ee;
        const float* Vp = cvals + (size_t)b * Cc * Ee;
        float m = -1e30f, lsum = 0.f, av = 0.f;
        for (int c = 0; c < Cc; ++c) {
            const float* kr = Kp + (size_t)c * Ee + h * 16;
            float s = 0.f;
            for (int j = 0; j < 16; ++j) s += sm_q[h * 16 + j] * kr[j];
            s *= 0.25f;                       // 1/sqrt(16)
            if (svcm[c]) s = NEGV;
            const float nm = fmaxf(m, s);
            const float sc = __expf(m - nm);
            const float p  = __expf(s - nm);
            lsum = lsum * sc + p;
            av   = av * sc + p * Vp[(size_t)c * Ee + h * 16 + d];
            m = nm;
        }
        glimpse[t] = av / lsum;
    }
    __syncthreads();
    {   // glimpse @ Wo^T + bo
        const float* wr = Wo + (size_t)t * Ee;
        float s = bo[t];
        for (int j = 0; j < Ee; ++j) s += glimpse[j] * wr[j];
        gproj[t] = s;
    }
    __syncthreads();
    if (t < Cc) {
        const float* pk = pkeys + (size_t)b * Cc * Ee + (size_t)t * Ee;
        float s = 0.f;
        for (int j = 0; j < Ee; ++j) s += gproj[j] * pk[j];
        s = tanhf(s * 0.0625f) * 10.0f;       // /sqrt(256), clip 10
        if (svcm[t]) s = NEGV;
        logits[t] = s;
    }
    __syncthreads();
    if (t == 0) {
        float mx = -1e30f; int arg = 0;
        for (int c = 0; c < Cc; ++c) if (logits[c] > mx) { mx = logits[c]; arg = c; }
        float se = 0.f;
        for (int c = 0; c < Cc; ++c) se += __expf(logits[c] - mx);
        red[0] = mx; red[1] = __logf(se); sarg = arg;
    }
    __syncthreads();

    const bool nb = is_new[b] != 0;
    if (t < Cc) out_prob[(size_t)b * Cc + t] = nb ? (logits[t] - red[0] - red[1]) : 0.0f;
    const int g = sarg;
    const float ge = nb ? cluster_emb[(size_t)b * Cc * Ee + (size_t)g * Ee + t] : 0.0f;
    out_gemb[(size_t)b * Ee + t] = ge;
    if (t == 0) out_guid[b] = nb ? (float)g : 0.0f;

    float* ab = aug + (size_t)b * 4 * Ee;
    ab[0 * Ee + t] = nb ? uclu[(size_t)b * Ee + t]  : 0.f;
    ab[1 * Ee + t] = nb ? cur[(size_t)b * Ee + t]   : 0.f;
    ab[2 * Ee + t] = ge;
    ab[3 * Ee + t] = nb ? depot[(size_t)b * Ee + t] : 0.f;
}

// ---------------------------------------------------------------------------
// S3: mha1 (4 queries x 500 keys, masked, online softmax) + Wo projection +
// residual + instance-norm over the 4-token axis  -> h1 [B,4,E]
// ---------------------------------------------------------------------------
__global__ __launch_bounds__(256)
void mha1_addnorm_kernel(const float* __restrict__ qf,    // [B,4,E]
                         const float* __restrict__ kf,    // [B,N,E]
                         const float* __restrict__ vf,    // [B,N,E]
                         const unsigned char* __restrict__ mask,
                         const unsigned char* __restrict__ cmask,
                         const float* __restrict__ Wo, const float* __restrict__ bo,
                         const float* __restrict__ aug,   // [B,4,E]
                         const float* __restrict__ g1, const float* __restrict__ b1,
                         float* __restrict__ h1)          // [B,4,E]
{
    const int b = blockIdx.x, t = threadIdx.x;
    __shared__ float sm_q[4 * Ee];
    __shared__ float sm_out[4 * Ee];
    for (int i = t; i < 4 * Ee; i += 256) sm_q[i] = qf[(size_t)b * 4 * Ee + i];
    __syncthreads();

    if (t < 64) {                       // (query, head) pairs
        const int q = t >> 4, h = t & 15;
        const float* Kp = kf + (size_t)b * Nn * Ee;
        const float* Vp = vf + (size_t)b * Nn * Ee;
        const unsigned char* mk = mask  + (size_t)b * Nn;
        const unsigned char* cm = cmask + (size_t)b * Nn;
        const float* qh = sm_q + q * Ee + h * 16;
        float m = -1e30f, lsum = 0.f, acc[16];
        for (int d = 0; d < 16; ++d) acc[d] = 0.f;
        for (int n = 0; n < Nn; ++n) {
            const float* kr = Kp + (size_t)n * Ee + h * 16;
            float s = 0.f;
            for (int j = 0; j < 16; ++j) s += qh[j] * kr[j];
            s *= 0.25f;
            if (mk[n] | cm[n]) s = NEGV;
            const float nm = fmaxf(m, s);
            const float sc = __expf(m - nm);
            const float p  = __expf(s - nm);
            lsum = lsum * sc + p;
            const float* vr = Vp + (size_t)n * Ee + h * 16;
            for (int d = 0; d < 16; ++d) acc[d] = acc[d] * sc + p * vr[d];
            m = nm;
        }
        const float inv = 1.0f / lsum;
        for (int d = 0; d < 16; ++d) sm_out[q * Ee + h * 16 + d] = acc[d] * inv;
    }
    __syncthreads();

    const int e = t;
    float a[4];
    for (int q = 0; q < 4; ++q) {
        const float* wr = Wo + (size_t)e * Ee;
        float s = bo[e];
        for (int j = 0; j < Ee; ++j) s += sm_out[q * Ee + j] * wr[j];
        a[q] = aug[(size_t)b * 4 * Ee + q * Ee + e] + s;
    }
    const float mu = 0.25f * (a[0] + a[1] + a[2] + a[3]);
    float var = 0.f;
    for (int q = 0; q < 4; ++q) { const float d = a[q] - mu; var += d * d; }
    var *= 0.25f;
    const float is = rsqrtf(var + 1e-5f);
    const float gg = g1[e], bb = b1[e];
    for (int q = 0; q < 4; ++q)
        h1[(size_t)b * 4 * Ee + q * Ee + e] = gg * (a[q] - mu) * is + bb;
}

// ---------------------------------------------------------------------------
// S4: add + instance-norm over 4-token axis (context path)
// ---------------------------------------------------------------------------
__global__ __launch_bounds__(256)
void addnorm4_kernel(const float* __restrict__ x1, const float* __restrict__ x2,
                     const float* __restrict__ g, const float* __restrict__ bb,
                     float* __restrict__ out)
{
    const int b = blockIdx.x, e = threadIdx.x;
    float a[4];
    for (int q = 0; q < 4; ++q)
        a[q] = x1[(size_t)b * 4 * Ee + q * Ee + e] + x2[(size_t)b * 4 * Ee + q * Ee + e];
    const float mu = 0.25f * (a[0] + a[1] + a[2] + a[3]);
    float var = 0.f;
    for (int q = 0; q < 4; ++q) { const float d = a[q] - mu; var += d * d; }
    var *= 0.25f;
    const float is = rsqrtf(var + 1e-5f);
    const float gg = g[e], b0 = bb[e];
    for (int q = 0; q < 4; ++q)
        out[(size_t)b * 4 * Ee + q * Ee + e] = gg * (a[q] - mu) * is + b0;
}

// ---------------------------------------------------------------------------
// S5: mha2 (N queries x 4 keys, no mask). 10 query rows per block; k/v in LDS.
// ---------------------------------------------------------------------------
#define NPB 10
__global__ __launch_bounds__(256)
void mha2_kernel(const float* __restrict__ q2,   // [B,N,E]
                 const float* __restrict__ k2,   // [B,4,E]
                 const float* __restrict__ v2,   // [B,4,E]
                 float* __restrict__ attn_out)   // [B,N,E]
{
    const int blk = blockIdx.x;
    const int b = blk / (Nn / NPB);
    const int nc = blk % (Nn / NPB);
    const int t = threadIdx.x;
    __shared__ float sk[4 * Ee], sv[4 * Ee], sq[Ee];
    for (int i = t; i < 4 * Ee; i += 256) {
        sk[i] = k2[(size_t)b * 4 * Ee + i];
        sv[i] = v2[(size_t)b * 4 * Ee + i];
    }
    __syncthreads();
    const int h = t >> 4;
    for (int ni = 0; ni < NPB; ++ni) {
        const int n = nc * NPB + ni;
        sq[t] = q2[(size_t)b * Nn * Ee + (size_t)n * Ee + t];
        __syncthreads();
        float s[4]; float mx = -1e30f;
        for (int j = 0; j < 4; ++j) {
            float d = 0.f;
            for (int x = 0; x < 16; ++x) d += sq[h * 16 + x] * sk[j * Ee + h * 16 + x];
            s[j] = d * 0.25f;
            mx = fmaxf(mx, s[j]);
        }
        float se = 0.f;
        for (int j = 0; j < 4; ++j) { s[j] = __expf(s[j] - mx); se += s[j]; }
        const float inv = 1.0f / se;
        float o = 0.f;
        for (int j = 0; j < 4; ++j) o += s[j] * sv[j * Ee + t];
        attn_out[(size_t)b * Nn * Ee + (size_t)n * Ee + t] = o * inv;
        __syncthreads();
    }
}

// ---------------------------------------------------------------------------
// S6: add + instance-norm over the N=500 axis (per (b,e) channel)
// ---------------------------------------------------------------------------
__global__ __launch_bounds__(256)
void addnormN_kernel(const float* __restrict__ x1, const float* __restrict__ x2,
                     const float* __restrict__ g, const float* __restrict__ bb,
                     float* __restrict__ out)
{
    const int b = blockIdx.x, e = threadIdx.x;
    const float* p1 = x1 + (size_t)b * Nn * Ee + e;
    const float* p2 = x2 + (size_t)b * Nn * Ee + e;
    float s = 0.f, ss = 0.f;
    for (int n = 0; n < Nn; ++n) {
        const float a = p1[(size_t)n * Ee] + p2[(size_t)n * Ee];
        s += a; ss += a * a;
    }
    const float invN = 1.0f / (float)Nn;
    const float mu = s * invN;
    const float var = ss * invN - mu * mu;
    const float is = rsqrtf(var + 1e-5f);
    const float gg = g[e], b0 = bb[e];
    float* po = out + (size_t)b * Nn * Ee + e;
    for (int n = 0; n < Nn; ++n) {
        const float a = p1[(size_t)n * Ee] + p2[(size_t)n * Ee];
        po[(size_t)n * Ee] = gg * (a - mu) * is + b0;
    }
}

// ---------------------------------------------------------------------------
// Host-side launch
// ---------------------------------------------------------------------------
static inline void launch_gemm(const float* A, const float* W, const float* bias,
                               float* C, int M, int K, int Nc, int relu,
                               hipStream_t s)
{
    const int tiles = (M / 16) * (Nc / 16);
    const int blocks = (tiles + 7) / 8;
    if (K == 256)      gemm_bf16_wmma<256><<<blocks, 256, 0, s>>>(A, W, bias, C, M, Nc, relu);
    else if (K == 512) gemm_bf16_wmma<512><<<blocks, 256, 0, s>>>(A, W, bias, C, M, Nc, relu);
    else               gemm_bf16_wmma<768><<<blocks, 256, 0, s>>>(A, W, bias, C, M, Nc, relu);
}

extern "C" void kernel_launch(void* const* d_in, const int* in_sizes, int n_in,
                              void* d_out, int out_size, void* d_ws, size_t ws_size,
                              hipStream_t stream)
{
    (void)in_sizes; (void)n_in; (void)out_size; (void)ws_size;

    // inputs (setup_inputs order)
    const float* depot      = (const float*)d_in[1];           // [B,1,E]
    const float* clu_emb    = (const float*)d_in[2];           // [B,C,E]
    const float* cur        = (const float*)d_in[3];           // [B,1,E]
    const float* node_emb   = (const float*)d_in[4];           // [B,N,E]
    const unsigned char* is_new = (const unsigned char*)d_in[6];   // [B]
    const unsigned char* cmask  = (const unsigned char*)d_in[7];   // [B,1,N]
    const unsigned char* vcm    = (const unsigned char*)d_in[8];   // [B,1,C]
    const unsigned char* mask   = (const unsigned char*)d_in[9];   // [B,1,N]
    // params (dict order, starting at index 12)
    const float* Wcq  = (const float*)d_in[12];
    const float* Wck  = (const float*)d_in[13];
    const float* Wcv  = (const float*)d_in[14];
    const float* Wcks = (const float*)d_in[15];
    const float* Wnq  = (const float*)d_in[16];
    const float* Wnv  = (const float*)d_in[17];
    const float* Wnk  = (const float*)d_in[18];
    const float* Waq  = (const float*)d_in[19];
    const float* Wav  = (const float*)d_in[20];
    const float* Wak  = (const float*)d_in[21];
    const float* mha_Wo  = (const float*)d_in[22];
    const float* mha_bo  = (const float*)d_in[23];
    const float* mha1_Wo = (const float*)d_in[24];
    const float* mha1_bo = (const float*)d_in[25];
    const float* mha2_Wo = (const float*)d_in[26];
    const float* mha2_bo = (const float*)d_in[27];
    const float* ff1_W1 = (const float*)d_in[28];
    const float* ff1_b1 = (const float*)d_in[29];
    const float* ff1_W2 = (const float*)d_in[30];
    const float* ff1_b2 = (const float*)d_in[31];
    const float* ff2_W1 = (const float*)d_in[32];
    const float* ff2_b1 = (const float*)d_in[33];
    const float* ff2_W2 = (const float*)d_in[34];
    const float* ff2_b2 = (const float*)d_in[35];
    const float* an1_g = (const float*)d_in[36];
    const float* an1_b = (const float*)d_in[37];
    const float* an2_g = (const float*)d_in[38];
    const float* an2_b = (const float*)d_in[39];
    const float* an3_g = (const float*)d_in[40];
    const float* an3_b = (const float*)d_in[41];
    const float* an4_g = (const float*)d_in[42];
    const float* an4_b = (const float*)d_in[43];

    // workspace layout (floats)
    const size_t S = (size_t)Bsz * Nn * Ee;       // 32,768,000
    float* ws = (float*)d_ws;
    float* buf0 = ws;                  // key_first   -> later ff2a (2S, with buf1)
    float* buf1 = ws + S;              // value_first
    float* buf2 = ws + 2 * S;          // q_second    -> later n1
    float* buf3 = ws + 3 * S;          // attn_out    -> later ff2b
    float* buf4 = ws + 4 * S;          // node_out (mha2 projection)
    float* sm   = ws + 5 * S;
    float* ctx3   = sm;                 sm += (size_t)Bsz * 768;
    float* uclu   = sm;                 sm += (size_t)Bsz * Ee;
    float* ckeys  = sm;                 sm += (size_t)Bsz * Cc * Ee;
    float* cvals  = sm;                 sm += (size_t)Bsz * Cc * Ee;
    float* pkeys  = sm;                 sm += (size_t)Bsz * Cc * Ee;
    float* cquery = sm;                 sm += (size_t)Bsz * Ee;
    float* aug    = sm;                 sm += (size_t)Bsz * 4 * Ee;
    float* qf     = sm;                 sm += (size_t)Bsz * 4 * Ee;
    float* h1     = sm;                 sm += (size_t)Bsz * 4 * Ee;
    float* ff1a   = sm;                 sm += (size_t)Bsz * 4 * FFd;
    float* ff1b   = sm;                 sm += (size_t)Bsz * 4 * Ee;
    float* uctx   = sm;                 sm += (size_t)Bsz * 4 * Ee;
    float* kk2    = sm;                 sm += (size_t)Bsz * 4 * Ee;
    float* vv2    = sm;                 sm += (size_t)Bsz * 4 * Ee;

    // output layout: [node_out3 | guidance_emb | cluster_guidance | clu_prob]
    float* out_nodes = (float*)d_out;
    float* out_gemb  = out_nodes + S;
    float* out_guid  = out_gemb + (size_t)Bsz * Ee;
    float* out_prob  = out_guid + Bsz;

    // ---- cluster-guidance branch ----
    build_ctx_kernel<<<Bsz, 256, 0, stream>>>(node_emb, cur, depot, mask, cmask,
                                              ctx3, uclu);
    launch_gemm(clu_emb, Wck,  nullptr, ckeys,  Bsz * Cc, Ee, Ee, 0, stream);
    launch_gemm(clu_emb, Wcv,  nullptr, cvals,  Bsz * Cc, Ee, Ee, 0, stream);
    launch_gemm(clu_emb, Wcks, nullptr, pkeys,  Bsz * Cc, Ee, Ee, 0, stream);
    launch_gemm(ctx3,    Wcq,  nullptr, cquery, Bsz, 3 * Ee, Ee, 0, stream);
    cluster_guidance_kernel<<<Bsz, 256, 0, stream>>>(
        cquery, ckeys, cvals, pkeys, vcm, is_new, mha_Wo, mha_bo,
        clu_emb, uclu, cur, depot, aug, out_gemb, out_guid, out_prob);

    // ---- node projections (big WMMA GEMMs) ----
    launch_gemm(node_emb, Wnv, nullptr, buf0, Bsz * Nn, Ee, Ee, 0, stream); // key_first
    launch_gemm(node_emb, Wnk, nullptr, buf1, Bsz * Nn, Ee, Ee, 0, stream); // value_first
    launch_gemm(node_emb, Wnq, nullptr, buf2, Bsz * Nn, Ee, Ee, 0, stream); // q_second
    launch_gemm(aug,      Waq, nullptr, qf,   Bsz * 4,  Ee, Ee, 0, stream); // q_first

    // ---- context <- nodes cross attention + FFN1 ----
    mha1_addnorm_kernel<<<Bsz, 256, 0, stream>>>(qf, buf0, buf1, mask, cmask,
                                                 mha1_Wo, mha1_bo, aug,
                                                 an1_g, an1_b, h1);
    launch_gemm(h1,   ff1_W1, ff1_b1, ff1a, Bsz * 4, Ee,  FFd, 1, stream);
    launch_gemm(ff1a, ff1_W2, ff1_b2, ff1b, Bsz * 4, FFd, Ee,  0, stream);
    addnorm4_kernel<<<Bsz, 256, 0, stream>>>(h1, ff1b, an2_g, an2_b, uctx);

    // ---- nodes <- context cross attention ----
    launch_gemm(uctx, Wav, nullptr, kk2, Bsz * 4, Ee, Ee, 0, stream);
    launch_gemm(uctx, Wak, nullptr, vv2, Bsz * 4, Ee, Ee, 0, stream);
    mha2_kernel<<<Bsz * (Nn / NPB), 256, 0, stream>>>(buf2, kk2, vv2, buf3);
    launch_gemm(buf3, mha2_Wo, mha2_bo, buf4, Bsz * Nn, Ee, Ee, 0, stream);

    // ---- norms + FFN2 (n1 into buf2; ff2a reuses buf0..buf1; ff2b -> buf3) ----
    addnormN_kernel<<<Bsz, 256, 0, stream>>>(node_emb, buf4, an3_g, an3_b, buf2);
    launch_gemm(buf2, ff2_W1, ff2_b1, buf0, Bsz * Nn, Ee,  FFd, 1, stream);
    launch_gemm(buf0, ff2_W2, ff2_b2, buf3, Bsz * Nn, FFd, Ee,  0, stream);
    addnormN_kernel<<<Bsz, 256, 0, stream>>>(buf2, buf3, an4_g, an4_b, out_nodes);
}
